// ReprogrammingLayer_m2_78675210928569
// MI455X (gfx1250) — matrix operand order repaired and checked by reading, not compiled, not run on hardware
//
#include <hip/hip_runtime.h>

typedef __attribute__((ext_vector_type(16))) _Float16 v16h;
typedef __attribute__((ext_vector_type(8)))  _Float16 v8h;
typedef __attribute__((ext_vector_type(8)))  float    v8f;
typedef __attribute__((ext_vector_type(4)))  unsigned u32x4;
typedef __attribute__((ext_vector_type(8)))  int      i32x8;
typedef __attribute__((ext_vector_type(4)))  int      i32x4;

#if defined(__AMDGCN__) && __has_builtin(__builtin_amdgcn_tensor_load_to_lds)
#define HAVE_TDM 1
#else
#define HAVE_TDM 0
#endif

__device__ __forceinline__ v8f wmma_f32_f16(v16h a, v16h b, v8f c) {
    // (neg_a, A, neg_b, B, c_mod, C, reuse_a, reuse_b)
    return __builtin_amdgcn_wmma_f32_16x16x32_f16(false, a, false, b, (short)0, c, false, false);
}

__device__ __forceinline__ void wait_tensorcnt0() {
#if defined(__AMDGCN__) && __has_builtin(__builtin_amdgcn_s_wait_tensorcnt)
    __builtin_amdgcn_s_wait_tensorcnt(0);
#elif defined(__AMDGCN__)
    asm volatile("s_wait_tensorcnt 0x0" ::: "memory");
#endif
}

#if HAVE_TDM
// TDM: DMA one 64x64 f16 tile (row stride in elements) from global into LDS.
// D# per CDNA5 ISA ch.8: group0 = {count=1 | lds_addr | global_addr | type=2},
// group1 = {data_size=2B, tensor dims (large; tiles never OOB), tile 64x64,
// dim0 stride}. Groups 2/3 zero (2D tile; tile_dim2/3 = 0 -> unused).
// This toolchain's builtin is the 6-arg form (g0, g1, g2, g3, g4, cpol).
__device__ __forceinline__ void tdm_load_tile_64x64(const _Float16* g, unsigned row_stride,
                                                    const _Float16* lds_dst) {
    unsigned long long ga = (unsigned long long)(uintptr_t)g;       // byte address
    unsigned la = (unsigned)(uintptr_t)lds_dst;                     // LDS byte offset
    u32x4 g0 = { 1u,                                                // count=1, user D#
                 la,                                                // [63:32] lds_addr
                 (unsigned)ga,                                      // [95:64] global lo
                 ((unsigned)(ga >> 32) & 0x01ffffffu) | 0x80000000u }; // addr hi | type=2
    i32x8 g1 = { (int)0x00010000,            // wg_mask=0, data_size=1 (2 bytes)
                 0,                          // atomic_barrier_addr = 0
                 (int)0x00000010,            // tensor_dim0 = 1<<20 (hi16), dim1 lo = 0
                 (int)((64u << 16) | 0x10u), // tile_dim0=64 | tensor_dim1 = 1<<20 (hi16)
                 64,                         // tile_dim1=64, tile_dim2=0
                 (int)row_stride,            // tensor_dim0_stride (lo32)
                 0, 0 };                     // stride hi, dim1_stride = 0
    i32x4 gz4 = { 0, 0, 0, 0 };
    i32x8 gz8 = { 0, 0, 0, 0, 0, 0, 0, 0 };
    __builtin_amdgcn_tensor_load_to_lds(g0, g1, gz4, gz4, gz8, 0);
}
#endif

// ---------------------------------------------------------------------------
// Weight transpose + f32->f16 convert:  Wt[n][k] = (f16) W[k][n],  1024x1024
// ---------------------------------------------------------------------------
__global__ __launch_bounds__(256) void wtrans_f16(const float* __restrict__ W,
                                                  _Float16* __restrict__ Wt) {
    int idx = blockIdx.x * blockDim.x + threadIdx.x;   // 0 .. 1024*1024-1
    int n = idx & 1023;
    int k = idx >> 10;
    Wt[(size_t)n * 1024 + k] = (_Float16)W[(size_t)k * 1024 + n];
}

// ---------------------------------------------------------------------------
// GEMM: Y = A[M,1024] @ W[1024,1024] + bias   (W given transposed f16: Wt[n][k])
// Wave computes a 32x32 tile (2x2 WMMA frags). Block = 8 waves = 128(M)x64(N).
// STORE_MODE: 0 = f16 [M,1024];  1 = f16 V-transpose [B][1024][4096];  2 = f32
// ---------------------------------------------------------------------------
template <bool A_IS_F32, int STORE_MODE>
__global__ __launch_bounds__(256) void gemm1024(const void* __restrict__ A,
                                                const _Float16* __restrict__ Wt,
                                                const float* __restrict__ bias,
                                                void* __restrict__ Y) {
    const int lane = threadIdx.x & 31;
    const int wid  = threadIdx.x >> 5;
    const int wm   = wid >> 1;
    const int wn   = wid & 1;
    const int m0   = blockIdx.y * 128 + wm * 32;
    const int n0   = blockIdx.x * 64  + wn * 32;
    const int half = lane >> 4;
    const int l16  = lane & 15;
    const int kbA  = half * 8;
    const int kbB  = half * 16;
    const int roff = half * 8;

    v8f acc[2][2] = {};

    for (int k0 = 0; k0 < 1024; k0 += 32) {
        v16h a[2], bm[2];
#pragma unroll
        for (int i = 0; i < 2; ++i) {
            int row = m0 + 16 * i + l16;
            if (A_IS_F32) {
                const float* p = (const float*)A + (size_t)row * 1024 + k0;
#pragma unroll
                for (int e = 0; e < 8; ++e) {
                    a[i][e]     = (_Float16)p[kbA + e];
                    a[i][8 + e] = (_Float16)p[16 + kbA + e];
                }
            } else {
                const _Float16* p = (const _Float16*)A + (size_t)row * 1024 + k0;
#pragma unroll
                for (int e = 0; e < 8; ++e) {
                    a[i][e]     = p[kbA + e];
                    a[i][8 + e] = p[16 + kbA + e];
                }
            }
        }
#pragma unroll
        for (int j = 0; j < 2; ++j) {
            int col = n0 + 16 * j + l16;
            const _Float16* p = Wt + (size_t)col * 1024 + k0;
#pragma unroll
            for (int e = 0; e < 16; ++e) bm[j][e] = p[e < 8 ? kbA + e : 8 + kbA + e];
        }
#pragma unroll
        for (int i = 0; i < 2; ++i)
#pragma unroll
            for (int j = 0; j < 2; ++j)
                acc[i][j] = wmma_f32_f16(a[i], bm[j], acc[i][j]);
    }

#pragma unroll
    for (int i = 0; i < 2; ++i)
#pragma unroll
        for (int j = 0; j < 2; ++j) {
            int col = n0 + 16 * j + l16;
            float bv = bias[col];
#pragma unroll
            for (int r = 0; r < 8; ++r) {
                int row = m0 + 16 * i + r + roff;
                float v = acc[i][j][r] + bv;
                if (STORE_MODE == 0) {
                    ((_Float16*)Y)[(size_t)row * 1024 + col] = (_Float16)v;
                } else if (STORE_MODE == 2) {
                    ((float*)Y)[(size_t)row * 1024 + col] = v;
                } else {
                    int b = row >> 12;            // S = 4096
                    int s = row & 4095;
                    ((_Float16*)Y)[((size_t)b * 1024 + col) * 4096 + s] = (_Float16)v;
                }
            }
        }
}

// ---------------------------------------------------------------------------
// Flash attention:  ctx[b,l,h,:] = softmax(q k^T / 8) v
// Q,K f16 [B*{L,S},1024] (head h = cols h*64..h*64+63); V f16 transposed
// [B][1024][S]. Block = 8 waves = 128 L rows for one (b,h); each wave owns 16
// rows. S streamed in 64-column steps; K/V tiles double-buffered in LDS,
// staged by the Tensor Data Mover (wave 0, TENSORcnt) and consumed by all 8
// waves. Per step: 8 WMMA (QK^T) + 8 WMMA (P.V). P goes C-layout -> LDS ->
// A-layout with an s_wait_dscnt fence.
// ---------------------------------------------------------------------------
__global__ __launch_bounds__(256) void flash_attn(const _Float16* __restrict__ Q,
                                                  const _Float16* __restrict__ K,
                                                  const _Float16* __restrict__ Vt,
                                                  _Float16* __restrict__ Ctx) {
    __shared__ _Float16 Kbuf[2][64][64];     // 16 KB, double-buffered K tile
    __shared__ _Float16 Vbuf[2][64][64];     // 16 KB, double-buffered V tile
    __shared__ _Float16 ldsP[8][16][64];     // 16 KB, per-wave P transpose tile

    const int lane = threadIdx.x & 31;
    const int wid  = threadIdx.x >> 5;
    const int half = lane >> 4;
    const int l16  = lane & 15;
    const int kbA  = half * 8;
    const int kbB  = half * 16;
    const int roff = half * 8;

    const int L = 2048, S = 4096, HD = 1024;
    const int bh = blockIdx.y;               // 0..31
    const int b  = bh >> 4;
    const int h  = bh & 15;
    const int lBase = blockIdx.x * 128 + wid * 16;
    const int hcol  = h * 64;

    const _Float16* Kb = K  + (size_t)b * S * HD;
    const _Float16* Vb = Vt + (size_t)b * 1024 * S;

    // Stage one 64-wide S block (K rows + V^T rows) into LDS buffer p.
    auto stage = [&](int s0, int p) {
#if HAVE_TDM
        if (wid == 0) {
            tdm_load_tile_64x64(Kb + (size_t)s0 * HD + hcol, HD, &Kbuf[p][0][0]);
            tdm_load_tile_64x64(Vb + (size_t)hcol * S + s0, S, &Vbuf[p][0][0]);
        }
#else
        int r = threadIdx.x >> 2;            // 0..63
        int c = (threadIdx.x & 3) * 16;      // 0,16,32,48
        const _Float16* ks = Kb + (size_t)(s0 + r) * HD + hcol + c;
        const _Float16* vs = Vb + (size_t)(hcol + r) * S + s0 + c;
        *(v8h*)&Kbuf[p][r][c]     = *(const v8h*)ks;
        *(v8h*)&Kbuf[p][r][c + 8] = *(const v8h*)(ks + 8);
        *(v8h*)&Vbuf[p][r][c]     = *(const v8h*)vs;
        *(v8h*)&Vbuf[p][r][c + 8] = *(const v8h*)(vs + 8);
#endif
    };

    // Q fragments for this wave's 16 rows: Dh=64 -> two K=32 chunks
    v16h aq[2];
    {
        const _Float16* p = Q + ((size_t)(b * L + lBase + l16)) * HD + hcol;
#pragma unroll
        for (int t = 0; t < 2; ++t)
#pragma unroll
            for (int e = 0; e < 8; ++e) {
                aq[t][e]     = p[32 * t + kbA + e];
                aq[t][8 + e] = p[32 * t + 16 + kbA + e];
            }
    }

    v8f o[4] = {};
    float mrow[8], lrow[8];
#pragma unroll
    for (int r = 0; r < 8; ++r) { mrow[r] = -__builtin_huge_valf(); lrow[r] = 0.0f; }

    const float scale = 0.125f;              // 1/sqrt(64)
    const int NIT = S / 64;                  // 64 iterations

    stage(0, 0);                             // prologue DMA into buffer 0

    for (int it = 0; it < NIT; ++it) {
        const int p = it & 1;
#if HAVE_TDM
        if (wid == 0) wait_tensorcnt0();     // buffer p landed in LDS
#endif
        __syncthreads();                     // all waves see buffer p; buffer 1-p free
        if (it + 1 < NIT) stage((it + 1) * 64, 1 - p);   // overlap next DMA

        // ---- scores: four 16x16 tiles over columns [it*64, it*64+64) ----
        v8f sc[4] = {};
#pragma unroll
        for (int jt = 0; jt < 4; ++jt) {
            const _Float16* kp = &Kbuf[p][16 * jt + l16][0];
#pragma unroll
            for (int t = 0; t < 2; ++t) {
                v16h bm;
#pragma unroll
                for (int e = 0; e < 16; ++e) bm[e] = kp[32 * t + (e < 8 ? kbA + e : 8 + kbA + e)];
                sc[jt] = wmma_f32_f16(aq[t], bm, sc[jt]);
            }
        }

        // ---- online softmax (row groups live in 16-lane halves) ----
#pragma unroll
        for (int r = 0; r < 8; ++r) {
            float sv[4], ev[4];
#pragma unroll
            for (int jt = 0; jt < 4; ++jt) sv[jt] = sc[jt][r] * scale;
            float mx = fmaxf(fmaxf(sv[0], sv[1]), fmaxf(sv[2], sv[3]));
            mx = fmaxf(mx, __shfl_xor(mx, 1));
            mx = fmaxf(mx, __shfl_xor(mx, 2));
            mx = fmaxf(mx, __shfl_xor(mx, 4));
            mx = fmaxf(mx, __shfl_xor(mx, 8));
            float mnew = fmaxf(mrow[r], mx);
            float corr = __expf(mrow[r] - mnew);
            float rs = 0.0f;
#pragma unroll
            for (int jt = 0; jt < 4; ++jt) { ev[jt] = __expf(sv[jt] - mnew); rs += ev[jt]; }
            rs += __shfl_xor(rs, 1);
            rs += __shfl_xor(rs, 2);
            rs += __shfl_xor(rs, 4);
            rs += __shfl_xor(rs, 8);
            lrow[r] = lrow[r] * corr + rs;
            mrow[r] = mnew;
#pragma unroll
            for (int t = 0; t < 4; ++t) o[t][r] *= corr;
#pragma unroll
            for (int jt = 0; jt < 4; ++jt)
                ldsP[wid][r + roff][16 * jt + l16] = (_Float16)ev[jt];
        }
        asm volatile("s_wait_dscnt 0" ::: "memory");   // intra-wave DS RAW fence

        // P (16x64) as two A-fragments
        v16h ap[2];
        {
            const _Float16* pp = &ldsP[wid][l16][0];
#pragma unroll
            for (int kc = 0; kc < 2; ++kc)
#pragma unroll
                for (int e = 0; e < 8; ++e) {
                    ap[kc][e]     = pp[32 * kc + kbA + e];
                    ap[kc][8 + e] = pp[32 * kc + 16 + kbA + e];
                }
        }

        // ---- o += P @ V  (4 column tiles of Dh x 2 K-chunks) ----
#pragma unroll
        for (int nt = 0; nt < 4; ++nt)
#pragma unroll
            for (int kc = 0; kc < 2; ++kc) {
                const _Float16* vp = &Vbuf[p][16 * nt + l16][32 * kc];
                v16h bv;
#pragma unroll
                for (int e = 0; e < 16; ++e) bv[e] = vp[e < 8 ? kbA + e : 8 + kbA + e];
                o[nt] = wmma_f32_f16(ap[kc], bv, o[nt]);
            }
    }

    // ---- normalize and store context (f16, row-major [B*L, 1024]) ----
#pragma unroll
    for (int r = 0; r < 8; ++r) {
        float inv = 1.0f / lrow[r];
        int row = lBase + r + roff;
        _Float16* cp = Ctx + ((size_t)(b * L + row)) * HD + hcol;
#pragma unroll
        for (int nt = 0; nt < 4; ++nt)
            cp[nt * 16 + l16] = (_Float16)(o[nt][r] * inv);
    }
}

// ---------------------------------------------------------------------------
// Host launcher
// ---------------------------------------------------------------------------
extern "C" void kernel_launch(void* const* d_in, const int* in_sizes, int n_in,
                              void* d_out, int out_size, void* d_ws, size_t ws_size,
                              hipStream_t stream) {
    const float* tgt = (const float*)d_in[0];   // [2,2048,1024]
    const float* src = (const float*)d_in[1];   // [2,4096,1024]
    const float* val = (const float*)d_in[2];   // [2,4096,1024]
    const float* Wq  = (const float*)d_in[3];
    const float* bq  = (const float*)d_in[4];
    const float* Wk  = (const float*)d_in[5];
    const float* bk  = (const float*)d_in[6];
    const float* Wv  = (const float*)d_in[7];
    const float* bv  = (const float*)d_in[8];
    const float* Wo  = (const float*)d_in[9];
    const float* bo  = (const float*)d_in[10];

    char* ws = (char*)d_ws;
    const size_t MB = 1024ull * 1024ull;
    _Float16* Wtq  = (_Float16*)(ws + 0 * MB);   // 2 MB each, pre-transposed
    _Float16* Wtk  = (_Float16*)(ws + 2 * MB);
    _Float16* Wtv  = (_Float16*)(ws + 4 * MB);
    _Float16* Wto  = (_Float16*)(ws + 6 * MB);
    _Float16* Q16  = (_Float16*)(ws + 8 * MB);   // [4096,1024] f16 : 8 MB
    _Float16* K16  = (_Float16*)(ws + 16 * MB);  // [8192,1024] f16 : 16 MB
    _Float16* Vt16 = (_Float16*)(ws + 32 * MB);  // [2][1024][4096] f16 : 16 MB
    _Float16* Ctx  = (_Float16*)(ws + 48 * MB);  // [4096,1024] f16 : 8 MB

    // 1) weight transpose+convert (4 x 1M elements)
    wtrans_f16<<<4096, 256, 0, stream>>>(Wq, Wtq);
    wtrans_f16<<<4096, 256, 0, stream>>>(Wk, Wtk);
    wtrans_f16<<<4096, 256, 0, stream>>>(Wv, Wtv);
    wtrans_f16<<<4096, 256, 0, stream>>>(Wo, Wto);

    // 2) projections (block tile 128x64; grid = (N/64, M/128))
    gemm1024<true, 0><<<dim3(16, 32), 256, 0, stream>>>(tgt, Wtq, bq, Q16);
    gemm1024<true, 0><<<dim3(16, 64), 256, 0, stream>>>(src, Wtk, bk, K16);
    gemm1024<true, 1><<<dim3(16, 64), 256, 0, stream>>>(val, Wtv, bv, Vt16);

    // 3) flash attention: grid = (L/128, B*H)
    flash_attn<<<dim3(16, 32), 256, 0, stream>>>(Q16, K16, Vt16, Ctx);

    // 4) output projection -> f32 d_out
    gemm1024<false, 2><<<dim3(16, 32), 256, 0, stream>>>(Ctx, Wto, bo, d_out);
}
